// L2MultiheadAttention_16630113370484
// MI455X (gfx1250) — compile-verified, matrix operand order checked by reading, not run on hardware
//
#include <hip/hip_runtime.h>

// Problem constants (from the reference): T=2048, N=2, D=1024, H=16, HD=64.
#define T_  2048
#define N_  2
#define D_  1024
#define H_  16
#define HD_ 64
#define M_  (T_ * N_)   // 4096 rows of flattened x

typedef __attribute__((ext_vector_type(16))) __bf16 v16bf;
typedef __attribute__((ext_vector_type(8)))  float  v8f;
typedef __attribute__((ext_vector_type(4)))  unsigned int u32x4;

union FragBF {
  v16bf v;
  u32x4 u[2];   // two 16-byte halves: elements 0..7 and 8..15
};

__device__ __forceinline__ v8f wmma_bf16(v16bf a, v16bf b, v8f c) {
  // 8 args: (neg_a, A, neg_b, B, c_mod, C, reuse_a, reuse_b)
  return __builtin_amdgcn_wmma_f32_16x16x32_bf16(false, a, false, b, (short)0, c,
                                                 false, false);
}

// --------------------------------------------------------------------------
// CDNA5 async global->LDS copy (ASYNCcnt-tracked), with sync fallback if the
// builtin is not declared by this toolchain.  The builtin takes
// (int4 addrspace(1)*, int4 addrspace(3)*, imm offset, imm cpol) per the
// compiler diagnostic from the previous round.
// --------------------------------------------------------------------------
#if defined(__has_builtin)
#  if __has_builtin(__builtin_amdgcn_global_load_async_to_lds_b128)
#    define ASYNC_LDS 1
#  endif
#endif
#ifndef ASYNC_LDS
#  define ASYNC_LDS 0
#endif

typedef __attribute__((__vector_size__(16))) int vint4;
typedef __attribute__((address_space(1))) vint4 as1_vint4;
typedef __attribute__((address_space(3))) vint4 as3_vint4;

__device__ __forceinline__ void async_cp16(const void* g, void* l) {
#if ASYNC_LDS
  __builtin_amdgcn_global_load_async_to_lds_b128((as1_vint4*)g, (as3_vint4*)l, 0, 0);
#else
  *(u32x4*)l = *(const u32x4*)g;
#endif
}

#if ASYNC_LDS
#  if __has_builtin(__builtin_amdgcn_s_wait_asynccnt)
#    define WAIT_ASYNC(n) __builtin_amdgcn_s_wait_asynccnt(n)
#  else
#    define WAIT_ASYNC(n) asm volatile("s_wait_asynccnt " #n ::: "memory")
#  endif
#else
#  define WAIT_ASYNC(n) do {} while (0)
#endif

// ---------------------------------------------------------------------------
// Prep kernels (tiny: <0.3 GFLOP combined)
// ---------------------------------------------------------------------------

__global__ __launch_bounds__(256)
void cvt_f32_bf16_kernel(const float* __restrict__ src, __bf16* __restrict__ dst, int n) {
  int i = blockIdx.x * 256 + threadIdx.x;
  if (i < n) dst[i] = (__bf16)src[i];
}

// dst[n][m] = (bf16) src[m][n] for 1024x1024 (weight pre-transpose so GEMM
// B-fragments are contiguous-in-K).
__global__ __launch_bounds__(256)
void transpose_to_bf16_kernel(const float* __restrict__ src, __bf16* __restrict__ dst) {
  int i = blockIdx.x * 256 + threadIdx.x;   // i = n*1024 + m
  int n = i >> 10, m = i & 1023;
  dst[i] = (__bf16)src[m * 1024 + n];
}

// Ch[h][e][d] = (1/sqrt(HD)) * sum_m qw[m,h,e] * vw[m,h,d]
__global__ __launch_bounds__(256)
void compute_C_kernel(const float* __restrict__ qw, const float* __restrict__ vw,
                      float* __restrict__ Ch) {
  const int h = blockIdx.x;
  const int tid = threadIdx.x;
  for (int it = 0; it < 16; ++it) {
    const int idx = tid * 16 + it;          // 0..4095
    const int e = idx >> 6, d = idx & 63;
    float acc = 0.f;
    for (int m = 0; m < D_; ++m)
      acc += qw[(m * H_ + h) * HD_ + e] * vw[(m * H_ + h) * HD_ + d];
    Ch[(h * HD_ + e) * HD_ + d] = acc * 0.125f;   // 1/sqrt(64)
  }
}

// Bt[(h*64+d)*1024 + m] = (bf16) sum_e qw[m,h,e] * Ch[h,e,d]   (transposed B)
__global__ __launch_bounds__(256)
void compute_Bt_kernel(const float* __restrict__ qw, const float* __restrict__ Ch,
                       __bf16* __restrict__ Bt) {
  const int i = blockIdx.x * 256 + threadIdx.x;   // i = n*1024 + m, n = h*64+d
  const int n = i >> 10, m = i & 1023;
  const int h = n >> 6, d = n & 63;
  float acc = 0.f;
  for (int e = 0; e < HD_; ++e)
    acc += qw[(m * H_ + h) * HD_ + e] * Ch[(h * HD_ + e) * HD_ + d];
  Bt[i] = (__bf16)acc;
}

// qsq[bh*T + t] = ||Q row||^2  (key norms for the reduced logit)
__global__ __launch_bounds__(256)
void qsq_kernel(const __bf16* __restrict__ Q, float* __restrict__ qsq) {
  const int i = blockIdx.x * 256 + threadIdx.x;   // 0 .. N*H*T-1
  const __bf16* p = Q + (size_t)i * HD_;
  float acc = 0.f;
  for (int d = 0; d < HD_; ++d) { float v = (float)p[d]; acc += v * v; }
  qsq[i] = acc;
}

// ---------------------------------------------------------------------------
// bf16 WMMA GEMM with async-LDS staging:
//   C[r][n] = sum_k A[r][k] * Bt[n][k],  M=4096, N=1024, K=1024.
// Block = 8 waves, tile 128x128; wave tile 32x64 (2x4 WMMA accumulators).
// A/B panels (128x32 each, 8 KB) double-buffered in LDS via
// global_load_async_to_lds_b128; fragments re-read with ds_load_b128 so each
// global byte feeds 8 waves (~64 FLOP per global byte).
// mode 0: write bf16 Q as [b][h][t][d]
// mode 1: write bf16 Y transposed as [b][h][d][t]
// mode 2: write f32 out[r][n] = val + bias[n]
// ---------------------------------------------------------------------------
__global__ __launch_bounds__(256)
void gemm_bf16_kernel(const __bf16* __restrict__ A, const __bf16* __restrict__ Bt,
                      float* __restrict__ outF, __bf16* __restrict__ outB,
                      const float* __restrict__ bias, int mode) {
  __shared__ __align__(16) __bf16 abuf[2][128 * 32];
  __shared__ __align__(16) __bf16 bbuf[2][128 * 32];

  const int lane = threadIdx.x & 31;
  const int wave = threadIdx.x >> 5;
  const int l16  = lane & 15;
  const int half = lane >> 4;
  const int wM = wave & 3, wN = wave >> 2;
  const int mblock = blockIdx.x * 128;
  const int nblock = blockIdx.y * 128;

  // Cooperative loader: each wave stages 16 rows (of 32 bf16 = 64 B) of A and B.
  const int ldRow = lane >> 2;        // 0..7
  const int ldSeg = (lane & 3) * 8;   // element offset within a 32-elem row

  auto issue = [&](int k0, int buf) {
#pragma unroll
    for (int sub = 0; sub < 2; ++sub) {
      const int r = wave * 16 + sub * 8 + ldRow;   // 0..127
      async_cp16(A + (size_t)(mblock + r) * 1024 + k0 + ldSeg,
                 &abuf[buf][r * 32 + ldSeg]);
      async_cp16(Bt + (size_t)(nblock + r) * 1024 + k0 + ldSeg,
                 &bbuf[buf][r * 32 + ldSeg]);
    }
  };

  const v8f zero8 = {0.f, 0.f, 0.f, 0.f, 0.f, 0.f, 0.f, 0.f};
  v8f acc[2][4];
#pragma unroll
  for (int i = 0; i < 2; ++i)
#pragma unroll
    for (int j = 0; j < 4; ++j) acc[i][j] = zero8;

  issue(0, 0);
  for (int c = 0; c < 32; ++c) {          // K = 32 chunks of 32
    const int buf = c & 1;
    if (c + 1 < 32) { issue((c + 1) * 32, buf ^ 1); WAIT_ASYNC(4); }
    else            { WAIT_ASYNC(0); }
    __syncthreads();

    FragBF a[2], b[4];
#pragma unroll
    for (int i = 0; i < 2; ++i) {
      const __bf16* p = &abuf[buf][(wM * 32 + i * 16 + l16) * 32 + half * 8];
      a[i].u[0] = *(const u32x4*)p;
      a[i].u[1] = *(const u32x4*)(p + 16);
    }
#pragma unroll
    for (int j = 0; j < 4; ++j) {
      const __bf16* p = &bbuf[buf][(wN * 64 + j * 16 + l16) * 32 + half * 8];
      b[j].u[0] = *(const u32x4*)p;
      b[j].u[1] = *(const u32x4*)(p + 16);
    }
#pragma unroll
    for (int i = 0; i < 2; ++i)
#pragma unroll
      for (int j = 0; j < 4; ++j)
        acc[i][j] = wmma_bf16(a[i].v, b[j].v, acc[i][j]);

    __syncthreads();   // all waves done reading before this buffer is refilled
  }

  // C layout: VGPR slot p, lane l -> row = p + 8*(l/16), col = l%16.
#pragma unroll
  for (int i = 0; i < 2; ++i)
#pragma unroll
    for (int j = 0; j < 4; ++j)
#pragma unroll
      for (int p = 0; p < 8; ++p) {
        const int row = mblock + wM * 32 + i * 16 + p + 8 * half;
        const int col = nblock + wN * 64 + j * 16 + l16;
        const float val = acc[i][j][p];
        if (mode == 2) {
          outF[(size_t)row * 1024 + col] = val + bias[col];
        } else {
          const int t = row >> 1, bb = row & 1;   // r = t*N + b, N==2
          const int h = col >> 6, d = col & 63;
          if (mode == 0)
            outB[((size_t)((bb * H_ + h) * T_ + t)) * HD_ + d] = (__bf16)val;
          else
            outB[((size_t)((bb * H_ + h) * HD_ + d)) * T_ + t] = (__bf16)val;
        }
      }
}

// ---------------------------------------------------------------------------
// Flash-attention with K = Q and V = Y.  All 8 waves of a block share one
// (b,h) and the same key stream, so the K-chunk (32x64) and Y-chunk (64x32,
// pre-transposed) are staged once per block via async global->LDS loads
// (double buffered).  Each wave owns 16 query rows; keys streamed 32 at a
// time.  logits = (2*q.k - ||k||^2)/sqrt(HD) (the ||q||^2 term cancels in the
// softmax).  Online softmax stats live per-VGPR-slot, uniform across each
// 16-lane group after xor-reductions.
// ---------------------------------------------------------------------------
__global__ __launch_bounds__(256)
void l2attn_kernel(const __bf16* __restrict__ Q, const __bf16* __restrict__ Yt,
                   const float* __restrict__ qsq, __bf16* __restrict__ Obuf) {
  __shared__ __align__(16) __bf16 kbuf[2][32 * 64];   // [s][d], 4 KB each
  __shared__ __align__(16) __bf16 ybuf[2][64 * 32];   // [d][s], 4 KB each
  __shared__ __align__(16) __bf16 plds[8][16 * 32];   // per-wave P transpose

  const int lane = threadIdx.x & 31;
  const int wave = threadIdx.x >> 5;
  const int l16  = lane & 15;
  const int half = lane >> 4;
  const int g  = blockIdx.x * 8 + wave;   // 4096 waves total
  const int bh = g >> 7;                  // same for all waves in the block
  const int qt = g & 127;
  const int b  = bh >> 4;
  const int h  = bh & 15;
  const int qbase = qt * 16;

  const __bf16* Qb = Q + (size_t)bh * T_ * HD_;    // [t][d]
  const __bf16* Yb = Yt + (size_t)bh * T_ * HD_;   // [d][t] (pre-transposed)
  const float*  qs = qsq + (size_t)bh * T_;

  // Cooperative chunk loader: K chunk is 4 KB contiguous; Y chunk is 64 rows
  // of 64 B.  One async b128 per wave for each.
  auto issue = [&](int s0, int buf) {
    async_cp16(Qb + (size_t)s0 * HD_ + wave * 256 + lane * 8,
               &kbuf[buf][wave * 256 + lane * 8]);
    const int row = wave * 8 + (lane >> 2);       // d row 0..63
    const int seg = (lane & 3) * 8;
    async_cp16(Yb + (size_t)row * T_ + s0 + seg,
               &ybuf[buf][row * 32 + seg]);
  };

  // Q A-fragments for this wave's 16 rows (2 chunks of K=32), loaded once.
  FragBF qa[2];
  {
    const __bf16* rp = Qb + (size_t)(qbase + l16) * HD_ + half * 8;
#pragma unroll
    for (int c = 0; c < 2; ++c) {
      qa[c].u[0] = *(const u32x4*)(rp + c * 32);
      qa[c].u[1] = *(const u32x4*)(rp + c * 32 + 16);
    }
  }

  const v8f zero8 = {0.f, 0.f, 0.f, 0.f, 0.f, 0.f, 0.f, 0.f};
  v8f o[4] = {zero8, zero8, zero8, zero8};   // 16x64 f32 accumulator
  float mrow[8], lrow[8];
#pragma unroll
  for (int j = 0; j < 8; ++j) { mrow[j] = -__builtin_inff(); lrow[j] = 0.f; }

  const float inv = 0.125f;   // 1/sqrt(64)

  issue(0, 0);
  for (int c = 0; c < T_ / 32; ++c) {
    const int buf = c & 1;
    const int s0 = c * 32;
    if (c + 1 < T_ / 32) { issue(s0 + 32, buf ^ 1); WAIT_ASYNC(2); }
    else                 { WAIT_ASYNC(0); }
    __syncthreads();

    // K^T B-fragments from LDS: B[k=d][n=s]; lane = key s, K contiguous in d.
    FragBF kb[2][2];
#pragma unroll
    for (int st = 0; st < 2; ++st) {
      const __bf16* rp = &kbuf[buf][(st * 16 + l16) * 64 + half * 8];
#pragma unroll
      for (int cc = 0; cc < 2; ++cc) {
        kb[st][cc].u[0] = *(const u32x4*)(rp + cc * 32);
        kb[st][cc].u[1] = *(const u32x4*)(rp + cc * 32 + 16);
      }
    }
    v8f sAcc[2];
#pragma unroll
    for (int st = 0; st < 2; ++st) {
      v8f z = zero8;
      z = wmma_bf16(qa[0].v, kb[st][0].v, z);
      z = wmma_bf16(qa[1].v, kb[st][1].v, z);
      sAcc[st] = z;
    }

    const float qsv0 = qs[s0 + l16];         // key norms, lane = key column
    const float qsv1 = qs[s0 + 16 + l16];
    float lg[2][8];
#pragma unroll
    for (int j = 0; j < 8; ++j) {
      lg[0][j] = inv * (2.f * sAcc[0][j] - qsv0);
      lg[1][j] = inv * (2.f * sAcc[1][j] - qsv1);
    }

    // Online softmax: row reductions across each 16-lane group.
    float pm[2][8];
#pragma unroll
    for (int j = 0; j < 8; ++j) {
      float v = fmaxf(lg[0][j], lg[1][j]);
      v = fmaxf(v, __shfl_xor(v, 1));
      v = fmaxf(v, __shfl_xor(v, 2));
      v = fmaxf(v, __shfl_xor(v, 4));
      v = fmaxf(v, __shfl_xor(v, 8));
      const float nm = fmaxf(mrow[j], v);
      const float sc = __expf(mrow[j] - nm);
      mrow[j] = nm;
      lrow[j] *= sc;
#pragma unroll
      for (int dt = 0; dt < 4; ++dt) o[dt][j] = o[dt][j] * sc;
      pm[0][j] = __expf(lg[0][j] - nm);
      pm[1][j] = __expf(lg[1][j] - nm);
      float rs = pm[0][j] + pm[1][j];
      rs += __shfl_xor(rs, 1);
      rs += __shfl_xor(rs, 2);
      rs += __shfl_xor(rs, 4);
      rs += __shfl_xor(rs, 8);
      lrow[j] += rs;
    }

    // P is in C-layout (lane = key column); bounce through per-wave LDS to
    // re-load it in A-fragment layout (lane = query row).
    __bf16* pw = &plds[wave][0];
#pragma unroll
    for (int st = 0; st < 2; ++st)
#pragma unroll
      for (int j = 0; j < 8; ++j) {
        const int row = j + 8 * half;
        const int col = st * 16 + l16;
        pw[row * 32 + col] = (__bf16)pm[st][j];
      }
    asm volatile("s_wait_dscnt 0x0" ::: "memory");   // stores visible wave-wide
    FragBF pa;
    {
      const __bf16* rp = pw + l16 * 32 + half * 8;
      pa.u[0] = *(const u32x4*)(rp);
      pa.u[1] = *(const u32x4*)(rp + 16);
    }
    asm volatile("s_wait_dscnt 0x0" ::: "memory");   // loads done before reuse

    // Y B-fragments from LDS: B[k=s][n=d] = ybuf[d][s]; K contiguous in s.
    FragBF yfr[4];
#pragma unroll
    for (int dt = 0; dt < 4; ++dt) {
      const __bf16* rp = &ybuf[buf][(dt * 16 + l16) * 32 + half * 8];
      yfr[dt].u[0] = *(const u32x4*)(rp);
      yfr[dt].u[1] = *(const u32x4*)(rp + 16);
    }
#pragma unroll
    for (int dt = 0; dt < 4; ++dt)
      o[dt] = wmma_bf16(pa.v, yfr[dt].v, o[dt]);

    __syncthreads();   // all waves done with this buffer before refill
  }

  // Normalize and store O as bf16 rows [r = t*N+b][h*64+d] for the final GEMM.
#pragma unroll
  for (int j = 0; j < 8; ++j) {
    const float rinv = 1.f / lrow[j];
    const int t = qbase + j + 8 * half;
#pragma unroll
    for (int dt = 0; dt < 4; ++dt) {
      const int d = dt * 16 + l16;
      Obuf[((size_t)(t * N_ + b)) * D_ + h * HD_ + d] = (__bf16)(o[dt][j] * rinv);
    }
  }
}

// ---------------------------------------------------------------------------
// Launch: prep -> Q/Y projection GEMMs -> key norms -> flash attention ->
// output projection.  Workspace layout (38.5 MiB, everything L2-resident):
//   [ 0,  8) xb bf16        [ 8, 10) WqT bf16      [10, 12) BmT bf16
//   [12, 20) Q bf16         [20, 28) Yt bf16       [28, 36) O bf16
//   [36, 38) out_w bf16     [38, +256K) qsq f32    [+256K, +512K) Ch f32
// ---------------------------------------------------------------------------
extern "C" void kernel_launch(void* const* d_in, const int* in_sizes, int n_in,
                              void* d_out, int out_size, void* d_ws, size_t ws_size,
                              hipStream_t stream) {
  (void)in_sizes; (void)n_in; (void)out_size; (void)ws_size;
  const float* x  = (const float*)d_in[0];
  const float* qw = (const float*)d_in[1];
  const float* vw = (const float*)d_in[2];
  const float* ow = (const float*)d_in[3];
  const float* ob = (const float*)d_in[4];
  float* out = (float*)d_out;

  char* ws = (char*)d_ws;
  __bf16* xb  = (__bf16*)(ws);                         // 8 MiB
  __bf16* WqT = (__bf16*)(ws + (size_t)(8u  << 20));   // 2 MiB
  __bf16* BmT = (__bf16*)(ws + (size_t)(10u << 20));   // 2 MiB
  __bf16* Qb  = (__bf16*)(ws + (size_t)(12u << 20));   // 8 MiB
  __bf16* YtB = (__bf16*)(ws + (size_t)(20u << 20));   // 8 MiB
  __bf16* Ob  = (__bf16*)(ws + (size_t)(28u << 20));   // 8 MiB
  __bf16* OwB = (__bf16*)(ws + (size_t)(36u << 20));   // 2 MiB
  float*  qsq = (float*) (ws + (size_t)(38u << 20));            // 256 KiB
  float*  Ch  = (float*) (ws + (size_t)(38u << 20) + (256u << 10)); // 256 KiB

  // Prep / precision conversion.
  cvt_f32_bf16_kernel<<<(M_ * D_) / 256, 256, 0, stream>>>(x, xb, M_ * D_);
  transpose_to_bf16_kernel<<<(D_ * D_) / 256, 256, 0, stream>>>(qw, WqT);
  cvt_f32_bf16_kernel<<<(D_ * D_) / 256, 256, 0, stream>>>(ow, OwB, D_ * D_);
  compute_C_kernel<<<H_, 256, 0, stream>>>(qw, vw, Ch);
  compute_Bt_kernel<<<(D_ * D_) / 256, 256, 0, stream>>>(qw, Ch, BmT);

  // Q = X Wq  (bf16, [b][h][t][d]);  Y = X B  (bf16, transposed [b][h][d][t]).
  dim3 ggrid(M_ / 128, D_ / 128);
  gemm_bf16_kernel<<<ggrid, 256, 0, stream>>>(xb, WqT, nullptr, Qb, nullptr, 0);
  gemm_bf16_kernel<<<ggrid, 256, 0, stream>>>(xb, BmT, nullptr, YtB, nullptr, 1);

  // Key norms.
  qsq_kernel<<<(N_ * H_ * T_) / 256, 256, 0, stream>>>(Qb, qsq);

  // Flash attention: 4096 waves (16 q-rows each) / 8 waves per block.
  l2attn_kernel<<<(N_ * H_ * (T_ / 16)) / 8, 256, 0, stream>>>(Qb, YtB, qsq, Ob);

  // Output projection: out = O @ out_w^T + b  (out_w is already [n][k]).
  gemm_bf16_kernel<<<ggrid, 256, 0, stream>>>(Ob, OwB, out, nullptr, ob, 2);
}
